// FrameTransformerEncoder_31190052504363
// MI455X (gfx1250) — compile-verified, matrix operand order checked by reading, not run on hardware
//
#include <hip/hip_runtime.h>
#include <hip/hip_bf16.h>

#define B_ 2
#define C_ 8
#define F_ 1024
#define W_ 512
#define NH_ 8
#define HD_ 128
#define CE_ 32  // C*EXP

typedef unsigned short u16;
typedef __attribute__((ext_vector_type(16))) __bf16 v16bf;
typedef __attribute__((ext_vector_type(8)))  float  v8f;

union FragU { uint4 q[2]; v16bf v; };

__device__ __forceinline__ u16 f2bf(float f) {
  union { float f; unsigned u; } v; v.f = f;
  unsigned u = v.u;
  unsigned r = u + 0x7FFFu + ((u >> 16) & 1u);
  return (u16)(r >> 16);
}
__device__ __forceinline__ float bf2f(u16 h) {
  union { unsigned u; float f; } v; v.u = ((unsigned)h) << 16;
  return v.f;
}

// CDNA5 async DMA: global -> LDS, 16B per lane, tracked by ASYNCcnt.
__device__ __forceinline__ void async_copy16(void* lds, const void* gmem) {
  unsigned off = (unsigned)(unsigned long long)lds;  // low 32 bits = LDS address
  asm volatile("global_load_async_to_lds_b128 %0, %1, off"
               :: "v"(off), "v"(gmem)
               : "memory");
}
template <int N>
__device__ __forceinline__ void wait_async() {
  asm volatile("s_wait_asynccnt %0" :: "i"(N) : "memory");
}

// Build a 16-bit A/B WMMA fragment from row-major storage following the CDNA5
// layout: lanes 0-15 hold rows row0..row0+15 with K {0..7,16..23}; lanes 16-31
// hold the same rows with K {8..15,24..31}.  Two b128 loads per lane.
__device__ __forceinline__ v16bf load_frag(const u16* base, int rowStride, int row0, int k0) {
  int lane = threadIdx.x & 31;
  int half = lane >> 4;
  const u16* p = base + (row0 + (lane & 15)) * rowStride + k0 + half * 8;
  FragU fu;
  fu.q[0] = *(const uint4*)(p);
  fu.q[1] = *(const uint4*)(p + 16);
  return fu.v;
}

// ---------------------------------------------------------------- cvt weights
__global__ void cvt_f32_bf16(const float* __restrict__ in, u16* __restrict__ out, long n) {
  long i = (long)blockIdx.x * blockDim.x + threadIdx.x;
  if (i < n) out[i] = f2bf(in[i]);
}

// --------------------------------------------------- frame norm -> bf16 (t,f)
__global__ void frame_norm_bf16T(const float* __restrict__ x, const float* __restrict__ nw,
                                 const float* __restrict__ nb, u16* __restrict__ outT) {
  int w  = blockIdx.x * blockDim.x + threadIdx.x;  // 0..511
  int bc = blockIdx.y;
  const float* xc = x + (long)bc * F_ * W_;
  float s = 0.f, ss = 0.f;
  for (int f = 0; f < F_; ++f) { float v = xc[(long)f * W_ + w]; s += v; ss += v * v; }
  float mu   = s * (1.f / F_);
  float var  = ss * (1.f / F_) - mu * mu;
  float rstd = rsqrtf(var + 1e-5f);
  const float* wr = nw + (long)(bc % C_) * F_;
  const float* br = nb + (long)(bc % C_) * F_;
  u16* o = outT + (long)bc * W_ * F_ + (long)w * F_;
  for (int f = 0; f < F_; ++f)
    o[f] = f2bf((xc[(long)f * W_ + w] - mu) * rstd * wr[f] + br[f]);
}

// --------------------------------------------------- frame norm -> f32 (f,w)
__global__ void frame_norm_f32(const float* __restrict__ x, const float* __restrict__ nw,
                               const float* __restrict__ nb, float* __restrict__ out) {
  int w  = blockIdx.x * blockDim.x + threadIdx.x;
  int bc = blockIdx.y;
  const float* xc = x + (long)bc * F_ * W_;
  float s = 0.f, ss = 0.f;
  for (int f = 0; f < F_; ++f) { float v = xc[(long)f * W_ + w]; s += v; ss += v * v; }
  float mu   = s * (1.f / F_);
  float var  = ss * (1.f / F_) - mu * mu;
  float rstd = rsqrtf(var + 1e-5f);
  const float* wr = nw + (long)(bc % C_) * F_;
  const float* br = nb + (long)(bc % C_) * F_;
  float* o = out + (long)bc * F_ * W_;
  for (int f = 0; f < F_; ++f)
    o[(long)f * W_ + w] = (xc[(long)f * W_ + w] - mu) * rstd * wr[f] + br[f];
}

// ------------------------------------------------------------ bf16 WMMA GEMM
// Y[z] = A[z%aMod] (MxK, row-major) @ B[z] given as Bt (NxK, row-major).
// 64x64 tile per 256-thread block (8 waves, each 16x32), K stepped by 32.
// Double-buffered LDS staging via async global->LDS DMA (ASYNCcnt).
template <bool TRANS_OUT, bool OUT_BF16, bool RESID>
__global__ void gemm_bf16_k(const u16* __restrict__ A, int lda, int aMod, long aStride,
                            const u16* __restrict__ Bt, int ldbt, long bStride,
                            void* __restrict__ Cout, int ldc, long cStride,
                            const float* __restrict__ Resid, long rStride, int K) {
  __shared__ __align__(16) u16 As[2][64 * 32];
  __shared__ __align__(16) u16 Bs[2][64 * 32];

  const int tid = threadIdx.x;
  const int z = blockIdx.z;
  const int tileM = blockIdx.y * 64;
  const int tileN = blockIdx.x * 64;

  const u16* Ab  = A  + (long)(z % aMod) * aStride;
  const u16* Btb = Bt + (long)z * bStride;

  const int wid = tid >> 5, lane = tid & 31;
  const int m0 = (wid & 3) * 16;
  const int n0 = (wid >> 2) * 32;

  v8f acc0 = {}, acc1 = {};

  const int srow = tid >> 2;
  const int scol = (tid & 3) * 8;
  const u16* gA = Ab  + (long)(tileM + srow) * lda  + scol;
  const u16* gB = Btb + (long)(tileN + srow) * ldbt + scol;
  u16* lA0 = &As[0][srow * 32 + scol];
  u16* lA1 = &As[1][srow * 32 + scol];
  u16* lB0 = &Bs[0][srow * 32 + scol];
  u16* lB1 = &Bs[1][srow * 32 + scol];

  async_copy16(lA0, gA);
  async_copy16(lB0, gB);

  const int nt = K / 32;
  int buf = 0;
  for (int i = 0; i < nt; ++i) {
    if (i + 1 < nt) {  // prefetch next tile into the other buffer
      async_copy16(buf ? lA0 : lA1, gA + (i + 1) * 32);
      async_copy16(buf ? lB0 : lB1, gB + (i + 1) * 32);
      wait_async<2>();  // current tile's 2 DMAs done; next 2 still in flight
    } else {
      wait_async<0>();
    }
    __syncthreads();
    const u16* Ac = As[buf];
    const u16* Bc = Bs[buf];
    v16bf a  = load_frag(Ac, 32, m0, 0);
    v16bf b0 = load_frag(Bc, 32, n0, 0);
    v16bf b1 = load_frag(Bc, 32, n0 + 16, 0);
    acc0 = __builtin_amdgcn_wmma_f32_16x16x32_bf16(false, a, false, b0, (short)0, acc0, false, false);
    acc1 = __builtin_amdgcn_wmma_f32_16x16x32_bf16(false, a, false, b1, (short)0, acc1, false, false);
    __syncthreads();  // compute done before this buffer is DMA'd again
    buf ^= 1;
  }

  const int nl = lane & 15;
  const int mb = (lane >> 4) * 8;

  if (TRANS_OUT) {  // write Yt (N rows, M cols) as bf16; lane owns 8 contiguous M
    u16* O = (u16*)Cout + (long)z * cStride;
#pragma unroll
    for (int j = 0; j < 2; ++j) {
      const v8f& a = j ? acc1 : acc0;
      long gn = tileN + n0 + j * 16 + nl;
      long gm = tileM + m0 + mb;
      union { u16 h[8]; uint4 q; } u;
#pragma unroll
      for (int r = 0; r < 8; ++r) u.h[r] = f2bf(a[r]);
      *(uint4*)(O + gn * ldc + gm) = u.q;
    }
  } else {
#pragma unroll
    for (int j = 0; j < 2; ++j) {
      const v8f& a = j ? acc1 : acc0;
#pragma unroll
      for (int r = 0; r < 8; ++r) {
        long idx = (long)(tileM + m0 + mb + r) * ldc + tileN + n0 + j * 16 + nl;
        float val = a[r];
        if (RESID) val += Resid[(long)z * rStride + idx];
        if (OUT_BF16) ((u16*)Cout + (long)z * cStride)[idx] = f2bf(val);
        else          ((float*)Cout + (long)z * cStride)[idx] = val;
      }
    }
  }
}

// ------------------------------------------------------------------- RoPE
__global__ void rope_inplace(u16* __restrict__ qT) {
  long i  = (long)blockIdx.x * blockDim.x + threadIdx.x;  // B*C*W*(F/2)
  int g2  = (int)(i & (F_ / 2 - 1));
  long tw = i >> 9;
  int t   = (int)(tw & (W_ - 1));
  long bc = tw >> 9;
  int g   = g2 * 2;
  int dd  = g & (HD_ - 1);
  float inv = __expf(-(float)dd * (9.210340371976184f / 128.f));  // 10000^(-dd/128)
  float ang = (float)t * inv;
  float cs = __cosf(ang), sn = __sinf(ang);
  unsigned* p = (unsigned*)qT + (((long)bc * W_ + t) * F_ + g) / 2;
  unsigned pk = *p;
  float x0 = bf2f((u16)pk), x1 = bf2f((u16)(pk >> 16));
  float y0 = x0 * cs - x1 * sn;
  float y1 = x1 * cs + x0 * sn;
  *p = (unsigned)f2bf(y0) | ((unsigned)f2bf(y1) << 16);
}

// --------------------------------------------------------------- attention
// One block per (b,c,head, 64 query rows). qT/kT/aT are (t,g); v is (g,t).
// All tile staging uses async global->LDS DMA.
__global__ void attention_k(const u16* __restrict__ qT, const u16* __restrict__ kT,
                            const u16* __restrict__ vv, u16* __restrict__ aT) {
  extern __shared__ __align__(16) char smem[];
  u16*   Qs = (u16*)smem;               // 64x128  16384B
  u16*   Ks = (u16*)(smem + 16384);     // 64x128  16384B
  float* Sm = (float*)(smem + 32768);   // 64x512 131072B
  u16*   Ps = (u16*)(smem + 163840);    // 64x512  65536B
  u16*   Vt = (u16*)(smem + 229376);    // 128x32   8192B
  __shared__ float red[64][4];

  const int tid = threadIdx.x;
  const int qt0 = blockIdx.x * 64;
  const int h   = blockIdx.y;
  const int bc  = blockIdx.z;

  const u16* qS = qT + (long)bc * W_ * F_ + (long)qt0 * F_ + h * HD_;
  const u16* kS = kT + (long)bc * W_ * F_ + h * HD_;
  const u16* vS = vv + (long)bc * F_ * W_ + (long)h * HD_ * W_;
  u16*       aS = aT + (long)bc * W_ * F_ + (long)qt0 * F_ + h * HD_;

#pragma unroll
  for (int it = 0; it < 4; ++it) {  // Q tile -> LDS (async DMA)
    int li = tid + it * 256;
    int row = li >> 4, c8 = (li & 15) * 8;
    async_copy16(&Qs[row * 128 + c8], qS + (long)row * F_ + c8);
  }

  const int wid = tid >> 5, lane = tid & 31;
  const int m0 = (wid & 3) * 16;
  const int nl = lane & 15, mb = (lane >> 4) * 8;
  const float scale = 0.03125f;  // 1/sqrt(F)

  for (int ktb = 0; ktb < 8; ++ktb) {  // S = Q @ K^T
    __syncthreads();  // prior reads of Ks complete before re-staging
#pragma unroll
    for (int it = 0; it < 4; ++it) {
      int li = tid + it * 256;
      int row = li >> 4, c8 = (li & 15) * 8;
      async_copy16(&Ks[row * 128 + c8], kS + (long)(ktb * 64 + row) * F_ + c8);
    }
    wait_async<0>();  // this wave's Q (first iter) + K DMAs done
    __syncthreads();
    int n0 = (wid >> 2) * 32;
    v8f s0 = {}, s1 = {};
#pragma unroll
    for (int kk = 0; kk < 128; kk += 32) {
      v16bf a  = load_frag(Qs, 128, m0, kk);
      v16bf b0 = load_frag(Ks, 128, n0, kk);
      v16bf b1 = load_frag(Ks, 128, n0 + 16, kk);
      s0 = __builtin_amdgcn_wmma_f32_16x16x32_bf16(false, a, false, b0, (short)0, s0, false, false);
      s1 = __builtin_amdgcn_wmma_f32_16x16x32_bf16(false, a, false, b1, (short)0, s1, false, false);
    }
#pragma unroll
    for (int r = 0; r < 8; ++r) {
      Sm[(m0 + mb + r) * 512 + ktb * 64 + n0 + nl]      = s0[r] * scale;
      Sm[(m0 + mb + r) * 512 + ktb * 64 + n0 + 16 + nl] = s1[r] * scale;
    }
  }
  __syncthreads();

  {  // softmax per row (4 threads x 128 cols per row)
    int row = tid & 63, q4 = tid >> 6;
    float* Sr = Sm + row * 512 + q4 * 128;
    float mx = -3.0e38f;
    for (int c = 0; c < 128; ++c) mx = fmaxf(mx, Sr[c]);
    red[row][q4] = mx;
    __syncthreads();
    float m = fmaxf(fmaxf(red[row][0], red[row][1]), fmaxf(red[row][2], red[row][3]));
    float s = 0.f;
    for (int c = 0; c < 128; ++c) { float e = __expf(Sr[c] - m); Sr[c] = e; s += e; }
    __syncthreads();
    red[row][q4] = s;
    __syncthreads();
    float inv = 1.f / (red[row][0] + red[row][1] + red[row][2] + red[row][3]);
    u16* Pr = Ps + row * 512 + q4 * 128;
    for (int c = 0; c < 128; ++c) Pr[c] = f2bf(Sr[c] * inv);
  }
  __syncthreads();

  // O = P @ V   (M=64 qt, N=128 d, K=512 kt)
  int n0 = (wid >> 2) * 64;
  v8f o[4] = {};
  for (int kt0 = 0; kt0 < 512; kt0 += 32) {
    __syncthreads();  // prior reads of Vt complete before re-staging
#pragma unroll
    for (int it = 0; it < 2; ++it) {
      int li = tid + it * 256;
      int row = li >> 2, c8 = (li & 3) * 8;
      async_copy16(&Vt[row * 32 + c8], vS + (long)row * W_ + kt0 + c8);
    }
    wait_async<0>();
    __syncthreads();
    v16bf a = load_frag(Ps, 512, m0, kt0);
#pragma unroll
    for (int j = 0; j < 4; ++j) {
      v16bf b = load_frag(Vt, 32, n0 + j * 16, 0);
      o[j] = __builtin_amdgcn_wmma_f32_16x16x32_bf16(false, a, false, b, (short)0, o[j], false, false);
    }
  }
#pragma unroll
  for (int j = 0; j < 4; ++j)
#pragma unroll
    for (int r = 0; r < 8; ++r)
      aS[(long)(m0 + mb + r) * F_ + n0 + j * 16 + nl] = f2bf(o[j][r]);
}

// -------------------------------------------------- conv1 3x3 + relu^2 -> bf16
__global__ void conv1_relu2(const float* __restrict__ y, const float* __restrict__ w,
                            u16* __restrict__ out) {
  long i  = (long)blockIdx.x * blockDim.x + threadIdx.x;  // B*CE*F*W
  int wo  = (int)(i & (W_ - 1));
  long r1 = i >> 9;
  int ho  = (int)(r1 & (F_ - 1));
  long r2 = r1 >> 10;
  int co  = (int)(r2 & (CE_ - 1));
  int b   = (int)(r2 >> 5);
  float acc = 0.f;
  for (int ci = 0; ci < C_; ++ci) {
    const float* yc = y + (long)(b * C_ + ci) * F_ * W_;
    const float* wk = w + (co * C_ + ci) * 9;
#pragma unroll
    for (int kh = 0; kh < 3; ++kh) {
      int hh = ho + kh - 1;
      if (hh < 0 || hh >= F_) continue;
#pragma unroll
      for (int kw = 0; kw < 3; ++kw) {
        int ww = wo + kw - 1;
        if (ww < 0 || ww >= W_) continue;
        acc += yc[(long)hh * W_ + ww] * wk[kh * 3 + kw];
      }
    }
  }
  float r = fmaxf(acc, 0.f);
  out[i] = f2bf(r * r);
}

// ------------------------------------ conv2 3x3 stride(2,1) + 1x1 ident -> out
__global__ void conv2_ident(const u16* __restrict__ c1, const float* __restrict__ w2,
                            const float* __restrict__ idw, const float* __restrict__ x2,
                            float* __restrict__ out) {
  long i  = (long)blockIdx.x * blockDim.x + threadIdx.x;  // B*C*512*512
  int wo  = (int)(i & (W_ - 1));
  long r1 = i >> 9;
  int ho  = (int)(r1 & 511);
  long r2 = r1 >> 9;
  int co  = (int)(r2 & (C_ - 1));
  int b   = (int)(r2 >> 3);
  float acc = 0.f;
  for (int ci = 0; ci < CE_; ++ci) {
    const u16* cc = c1 + (long)(b * CE_ + ci) * F_ * W_;
    const float* wk = w2 + (co * CE_ + ci) * 9;
#pragma unroll
    for (int kh = 0; kh < 3; ++kh) {
      int hh = 2 * ho + kh - 1;
      if (hh < 0 || hh >= F_) continue;
#pragma unroll
      for (int kw = 0; kw < 3; ++kw) {
        int ww = wo + kw - 1;
        if (ww < 0 || ww >= W_) continue;
        acc += bf2f(cc[(long)hh * W_ + ww]) * wk[kh * 3 + kw];
      }
    }
  }
  for (int ci = 0; ci < C_; ++ci)
    acc += idw[co * C_ + ci] * x2[((long)(b * C_ + ci) * F_ + 2 * ho) * W_ + wo];
  out[i] = acc;
}

// ---------------------------------------------------------------------- host
extern "C" void kernel_launch(void* const* d_in, const int* in_sizes, int n_in,
                              void* d_out, int out_size, void* d_ws, size_t ws_size,
                              hipStream_t stream) {
  (void)in_sizes; (void)n_in; (void)out_size; (void)ws_size;
  const float* x   = (const float*)d_in[0];
  const float* n1w = (const float*)d_in[1];
  const float* n1b = (const float*)d_in[2];
  const float* wq  = (const float*)d_in[3];
  const float* wk  = (const float*)d_in[4];
  const float* wv  = (const float*)d_in[5];
  const float* wo  = (const float*)d_in[6];
  const float* n2w = (const float*)d_in[7];
  const float* n2b = (const float*)d_in[8];
  const float* c1w = (const float*)d_in[9];
  const float* c2w = (const float*)d_in[10];
  const float* idw = (const float*)d_in[11];
  float* out = (float*)d_out;

  // ws layout (bytes); lifetime-aliased regions; peak 150,994,944 B
  char* ws = (char*)d_ws;
  const long MB16 = 16777216L;
  u16*   wqb = (u16*)(ws + 0 * MB16);
  u16*   wkb = (u16*)(ws + 1 * MB16);
  u16*   wvb = (u16*)(ws + 2 * MB16);
  u16*   wob = (u16*)(ws + 3 * MB16);
  u16*   xnT = (u16*)(ws + 4 * MB16);
  u16*   qTb = (u16*)(ws + 5 * MB16);
  u16*   kTb = (u16*)(ws + 6 * MB16);
  u16*   vb  = (u16*)(ws + 7 * MB16);
  u16*   aTb = (u16*)(ws + 8 * MB16);
  float* x2  = (float*)(ws + 0 * MB16);  // over dead wq/wk bf16
  float* y2  = (float*)(ws + 2 * MB16);  // over dead wv/wo bf16
  u16*   c1  = (u16*)(ws + 4 * MB16);    // over dead xnT/qT/kT/v

  const long nw = (long)C_ * F_ * F_;  // 8,388,608
  cvt_f32_bf16<<<(unsigned)(nw / 256), 256, 0, stream>>>(wq, wqb, nw);
  cvt_f32_bf16<<<(unsigned)(nw / 256), 256, 0, stream>>>(wk, wkb, nw);
  cvt_f32_bf16<<<(unsigned)(nw / 256), 256, 0, stream>>>(wv, wvb, nw);
  cvt_f32_bf16<<<(unsigned)(nw / 256), 256, 0, stream>>>(wo, wob, nw);

  frame_norm_bf16T<<<dim3(W_ / 256, B_ * C_), 256, 0, stream>>>(x, n1w, n1b, xnT);

  dim3 gg(W_ / 64, F_ / 64, B_ * C_);  // (8,16,16)
  gemm_bf16_k<true, true, false><<<gg, 256, 0, stream>>>(
      wqb, F_, C_, (long)F_ * F_, xnT, F_, (long)W_ * F_, qTb, F_, (long)W_ * F_, nullptr, 0, F_);
  gemm_bf16_k<true, true, false><<<gg, 256, 0, stream>>>(
      wkb, F_, C_, (long)F_ * F_, xnT, F_, (long)W_ * F_, kTb, F_, (long)W_ * F_, nullptr, 0, F_);
  gemm_bf16_k<false, true, false><<<gg, 256, 0, stream>>>(
      wvb, F_, C_, (long)F_ * F_, xnT, F_, (long)W_ * F_, vb, W_, (long)F_ * W_, nullptr, 0, F_);

  const long nr = (long)B_ * C_ * W_ * (F_ / 2);  // 4,194,304
  rope_inplace<<<(unsigned)(nr / 256), 256, 0, stream>>>(qTb);
  rope_inplace<<<(unsigned)(nr / 256), 256, 0, stream>>>(kTb);

  attention_k<<<dim3(W_ / 64, NH_, B_ * C_), 256, 237568, stream>>>(qTb, kTb, vb, aTb);

  gemm_bf16_k<false, false, true><<<gg, 256, 0, stream>>>(
      wob, F_, C_, (long)F_ * F_, aTb, F_, (long)W_ * F_, x2, W_, (long)F_ * W_, x, (long)F_ * W_, F_);

  frame_norm_f32<<<dim3(W_ / 256, B_ * C_), 256, 0, stream>>>(x2, n2w, n2b, y2);

  const long nc1 = (long)B_ * CE_ * F_ * W_;  // 33,554,432
  conv1_relu2<<<(unsigned)(nc1 / 256), 256, 0, stream>>>(y2, c1w, c1);

  const long nc2 = (long)B_ * C_ * (F_ / 2) * W_;  // 4,194,304
  conv2_ident<<<(unsigned)(nc2 / 256), 256, 0, stream>>>(c1, c2w, idw, x2, out);
}
